// Attention_41102837022722
// MI455X (gfx1250) — matrix operand order repaired and checked by reading, not compiled, not run on hardware
//
#include <hip/hip_runtime.h>

typedef __attribute__((ext_vector_type(16))) _Float16 v16h;
typedef __attribute__((ext_vector_type(8)))  _Float16 v8h;
typedef __attribute__((ext_vector_type(8)))  float    v8f;

#define DIM   2048
#define S_LEN 2048
#define NH    32
#define NKV   8
#define HD    64

static __device__ __forceinline__ v8f wmma_f16(v16h a, v16h b, v8f c) {
  // 8 args: (neg_a, A, neg_b, B, c_mod, C, reuse_a, reuse_b)
  return __builtin_amdgcn_wmma_f32_16x16x32_f16(false, a, false, b, (short)0, c,
                                                false, false);
}

// Async copy 16B/lane global -> LDS (CDNA5 async engine, tracked by ASYNCcnt).
static __device__ __forceinline__ void async_ld_b128(unsigned lds_off,
                                                     const _Float16* gptr) {
  asm volatile("global_load_async_to_lds_b128 %0, %1, off"
               :: "v"(lds_off), "v"((unsigned long long)(uintptr_t)gptr)
               : "memory");
}
static __device__ __forceinline__ void wait_async0() {
  asm volatile("s_wait_asynccnt 0x0" ::: "memory");
}

// ---------------------------------------------------------------------------
// Kernel 0: one-time f32 -> f16 conversion (bandwidth-bound).
// ---------------------------------------------------------------------------
__global__ __launch_bounds__(256) void cvt_f16_kernel(
    const float* __restrict__ in, _Float16* __restrict__ out, int n)
{
  int i = (blockIdx.x * 256 + threadIdx.x) * 8;
  if (i < n) {
    const float4* p = (const float4*)(in + i);
    float4 u = p[0], w = p[1];
    v8h h;
    h[0] = (_Float16)u.x; h[1] = (_Float16)u.y;
    h[2] = (_Float16)u.z; h[3] = (_Float16)u.w;
    h[4] = (_Float16)w.x; h[5] = (_Float16)w.y;
    h[6] = (_Float16)w.z; h[7] = (_Float16)w.w;
    *(v8h*)(out + i) = h;
  }
}

// ---------------------------------------------------------------------------
// Kernel 1: y = x @ W^T (f16 WMMA, f32 accum) + fused RoPE epilogue.
// Block = 4 waves on one head: the shared 64x32 B tile is staged to LDS once
// per block with async loads (double-buffered); each wave runs a 16x64 tile
// (4 WMMAs per K-step sharing one A fragment).
// ---------------------------------------------------------------------------
__global__ __launch_bounds__(128) void qkv_rope_kernel(
    const _Float16* __restrict__ x, const _Float16* __restrict__ w,
    _Float16* __restrict__ out, int n_heads, int do_rope, float scale)
{
  __shared__ _Float16 btile[2][64][40];   // 80B row stride: 16B-aligned, no bank conflicts
  __shared__ float ctile[4][16][65];
  const int t    = threadIdx.x;
  const int lane = t & 31;
  const int wv   = t >> 5;
  const int half = lane >> 4;
  const int l15  = lane & 15;
  const int m0   = blockIdx.x * 64 + wv * 16;
  const int h    = blockIdx.y;
  const int arow = m0 + l15;

  // Stage B tile (rows h*64..h*64+63, k kk..kk+31) via async engine:
  // 256 x 16B chunks, 2 per thread.
  auto stage = [&](int buf, int kk) {
#pragma unroll
    for (int i = 0; i < 2; ++i) {
      int idx = t + i * 128;
      int row = idx >> 2, ch = idx & 3;
      async_ld_b128((unsigned)(uintptr_t)&btile[buf][row][ch * 8],
                    &w[(size_t)(h * HD + row) * DIM + kk + ch * 8]);
    }
  };
  stage(0, 0);

  v8f c0 = {}, c1 = {}, c2 = {}, c3 = {};
  for (int kk = 0; kk < DIM; kk += 32) {
    const int buf = (kk >> 5) & 1;
    wait_async0();
    __syncthreads();                      // staged tile visible block-wide
    if (kk + 32 < DIM) stage(buf ^ 1, kk + 32);

    __builtin_prefetch(&x[(size_t)arow * DIM + kk + 128], 0, 0);
    v16h a, b0, b1, b2, b3;
#pragma unroll
    for (int e = 0; e < 16; ++e) {
      int ka = kk + ((e < 8) ? e : e + 8) + 8 * half;  // A-fragment K map
      a[e] = x[(size_t)arow * DIM + ka];
      int kb = ((e < 8) ? e : e + 8) + 8 * half;       // B-fragment K map
      b0[e] = btile[buf][ 0 + l15][kb];
      b1[e] = btile[buf][16 + l15][kb];
      b2[e] = btile[buf][32 + l15][kb];
      b3[e] = btile[buf][48 + l15][kb];
    }
    c0 = wmma_f16(a, b0, c0);
    c1 = wmma_f16(a, b1, c1);
    c2 = wmma_f16(a, b2, c2);
    c3 = wmma_f16(a, b3, c3);
  }
#pragma unroll
  for (int r = 0; r < 8; ++r) {
    int row = r + 8 * half;
    ctile[wv][row][ 0 + l15] = c0[r];
    ctile[wv][row][16 + l15] = c1[r];
    ctile[wv][row][32 + l15] = c2[r];
    ctile[wv][row][48 + l15] = c3[r];
  }
  asm volatile("s_wait_dscnt 0x0" ::: "memory");  // per-wave LDS turnaround

  // Per-wave RoPE epilogue: 32 lanes cover 16 rows x 32 pairs.
  const int mr = lane >> 1;
  const int pbase = (lane & 1) * 16;
  const int grow = m0 + mr;
  const int b = grow >> 11;            // / S_LEN
  const int s = grow & (S_LEN - 1);
  _Float16* orow = out + (((size_t)b * n_heads + h) * S_LEN + s) * HD;
#pragma unroll
  for (int j = 0; j < 16; ++j) {
    int p = pbase + j;
    float x1 = ctile[wv][mr][2 * p];
    float x2 = ctile[wv][mr][2 * p + 1];
    float r1, r2;
    if (do_rope) {
      // inv_freq = 10000^(-p/32) = exp(-p * ln(1e4)/32)
      float ang = (float)s * __expf(-(float)p * 0.28782313662425572f);
      float cs = __cosf(ang), sn = __sinf(ang);
      r1 = x1 * cs - x2 * sn;
      r2 = x1 * sn + x2 * cs;
    } else { r1 = x1; r2 = x2; }
    orow[2 * p]     = (_Float16)(r1 * scale);
    orow[2 * p + 1] = (_Float16)(r2 * scale);
  }
}

// ---------------------------------------------------------------------------
// Kernel 2: causal flash attention (GQA). One wave per 16 query rows.
// Scores + P.V via v_wmma_f32_16x16x32_f16; online softmax in f32.
// (Per-wave loads: divergent kv trip counts forbid block barriers here.)
// ---------------------------------------------------------------------------
__global__ __launch_bounds__(128) void attn_kernel(
    const _Float16* __restrict__ q, const _Float16* __restrict__ k,
    const _Float16* __restrict__ v, _Float16* __restrict__ o)
{
  __shared__ _Float16 p_lds[4][16][32];
  const int lane = threadIdx.x & 31;
  const int wv   = threadIdx.x >> 5;
  const int half = lane >> 4;
  const int l15  = lane & 15;
  const int b    = blockIdx.z;
  const int h    = blockIdx.y;
  const int kvh  = h >> 2;                   // N_REP = 4
  const int q0   = (blockIdx.x * 4 + wv) * 16;
  const _Float16* qp = q + ((size_t)b * NH  + h)   * S_LEN * HD;
  const _Float16* kp = k + ((size_t)b * NKV + kvh) * S_LEN * HD;
  const _Float16* vp = v + ((size_t)b * NKV + kvh) * S_LEN * HD;

  const int qr = q0 + l15;
  v16h a0, a1;
#pragma unroll
  for (int e = 0; e < 16; ++e) {
    int d = ((e < 8) ? e : e + 8) + 8 * half;
    a0[e] = qp[(size_t)qr * HD + d];
    a1[e] = qp[(size_t)qr * HD + 32 + d];
  }

  v8f o0 = {}, o1 = {}, o2 = {}, o3 = {};
  float mst[8], lst[8];
#pragma unroll
  for (int r = 0; r < 8; ++r) { mst[r] = -3.0e38f; lst[r] = 0.0f; }

  const int kv_end = q0 + 16;                // causal bound (exclusive)
  for (int kk = 0; kk < kv_end; kk += 32) {
    v16h b00, b01, b10, b11;
#pragma unroll
    for (int e = 0; e < 16; ++e) {
      int kd = ((e < 8) ? e : e + 8) + 8 * half;
      b00[e] = kp[(size_t)(kk + l15) * HD + kd];
      b01[e] = kp[(size_t)(kk + l15) * HD + 32 + kd];
      b10[e] = kp[(size_t)(kk + 16 + l15) * HD + kd];
      b11[e] = kp[(size_t)(kk + 16 + l15) * HD + 32 + kd];
    }
    v8f s0 = {}, s1 = {};
    s0 = wmma_f16(a0, b00, s0);
    s0 = wmma_f16(a1, b01, s0);
    s1 = wmma_f16(a0, b10, s1);
    s1 = wmma_f16(a1, b11, s1);

    // causal mask + online softmax (branch-free; EXEC stays all-ones)
#pragma unroll
    for (int r = 0; r < 8; ++r) {
      int qrow = q0 + r + 8 * half;
      s0[r] = (kk + l15      > qrow) ? -3.0e38f : s0[r];
      s1[r] = (kk + 16 + l15 > qrow) ? -3.0e38f : s1[r];
      float mx = fmaxf(s0[r], s1[r]);
      mx = fmaxf(mx, __shfl_xor(mx, 1));
      mx = fmaxf(mx, __shfl_xor(mx, 2));
      mx = fmaxf(mx, __shfl_xor(mx, 4));
      mx = fmaxf(mx, __shfl_xor(mx, 8));     // reduce within 16-lane half
      float mnew  = fmaxf(mst[r], mx);
      float alpha = __expf(mst[r] - mnew);
      mst[r] = mnew;
      float p0 = __expf(s0[r] - mnew);
      float p1 = __expf(s1[r] - mnew);
      float rs = p0 + p1;
      rs += __shfl_xor(rs, 1);
      rs += __shfl_xor(rs, 2);
      rs += __shfl_xor(rs, 4);
      rs += __shfl_xor(rs, 8);
      lst[r] = lst[r] * alpha + rs;
      o0[r] *= alpha; o1[r] *= alpha; o2[r] *= alpha; o3[r] *= alpha;
      int m = r + 8 * half;
      p_lds[wv][m][l15]      = (_Float16)p0;
      p_lds[wv][m][16 + l15] = (_Float16)p1;
    }
    asm volatile("s_wait_dscnt 0x0" ::: "memory");

    // P as A-fragment (16x32), V as four B-fragments
    v16h pa;
#pragma unroll
    for (int e = 0; e < 16; ++e) {
      int kc = ((e < 8) ? e : e + 8) + 8 * half;
      pa[e] = p_lds[wv][l15][kc];
    }
    v16h vb0, vb1, vb2, vb3;
#pragma unroll
    for (int e = 0; e < 16; ++e) {
      int kr = kk + ((e < 8) ? e : e + 8) + 8 * half;
      vb0[e] = vp[(size_t)kr * HD + l15];
      vb1[e] = vp[(size_t)kr * HD + 16 + l15];
      vb2[e] = vp[(size_t)kr * HD + 32 + l15];
      vb3[e] = vp[(size_t)kr * HD + 48 + l15];
    }
    o0 = wmma_f16(pa, vb0, o0);
    o1 = wmma_f16(pa, vb1, o1);
    o2 = wmma_f16(pa, vb2, o2);
    o3 = wmma_f16(pa, vb3, o3);
  }

  // epilogue: normalize and store f16 [B][S][H*HD] for the output GEMM
#pragma unroll
  for (int r = 0; r < 8; ++r) {
    int m = q0 + r + 8 * half;
    float inv = 1.0f / lst[r];
    size_t base = ((size_t)b * S_LEN + m) * DIM + h * HD;
    o[base + l15]      = (_Float16)(o0[r] * inv);
    o[base + 16 + l15] = (_Float16)(o1[r] * inv);
    o[base + 32 + l15] = (_Float16)(o2[r] * inv);
    o[base + 48 + l15] = (_Float16)(o3[r] * inv);
  }
}

// ---------------------------------------------------------------------------
// Kernel 3: out = attn @ wo^T (f16 WMMA, f32 result). Same async-staged
// B-tile scheme as kernel 1; block = 4 waves sharing one 64-col n-block.
// ---------------------------------------------------------------------------
__global__ __launch_bounds__(128) void oproj_kernel(
    const _Float16* __restrict__ a, const _Float16* __restrict__ w,
    float* __restrict__ out)
{
  __shared__ _Float16 btile[2][64][40];
  const int t    = threadIdx.x;
  const int lane = t & 31;
  const int wv   = t >> 5;
  const int half = lane >> 4;
  const int l15  = lane & 15;
  const int n0   = blockIdx.x * 64;
  const int m0   = blockIdx.y * 64 + wv * 16;
  const int arow = m0 + l15;

  auto stage = [&](int buf, int kk) {
#pragma unroll
    for (int i = 0; i < 2; ++i) {
      int idx = t + i * 128;
      int row = idx >> 2, ch = idx & 3;
      async_ld_b128((unsigned)(uintptr_t)&btile[buf][row][ch * 8],
                    &w[(size_t)(n0 + row) * DIM + kk + ch * 8]);
    }
  };
  stage(0, 0);

  v8f c0 = {}, c1 = {}, c2 = {}, c3 = {};
  for (int kk = 0; kk < DIM; kk += 32) {
    const int buf = (kk >> 5) & 1;
    wait_async0();
    __syncthreads();
    if (kk + 32 < DIM) stage(buf ^ 1, kk + 32);

    __builtin_prefetch(&a[(size_t)arow * DIM + kk + 128], 0, 0);
    v16h af, b0, b1, b2, b3;
#pragma unroll
    for (int e = 0; e < 16; ++e) {
      int ka = kk + ((e < 8) ? e : e + 8) + 8 * half;
      af[e] = a[(size_t)arow * DIM + ka];
      int kb = ((e < 8) ? e : e + 8) + 8 * half;
      b0[e] = btile[buf][ 0 + l15][kb];
      b1[e] = btile[buf][16 + l15][kb];
      b2[e] = btile[buf][32 + l15][kb];
      b3[e] = btile[buf][48 + l15][kb];
    }
    c0 = wmma_f16(af, b0, c0);
    c1 = wmma_f16(af, b1, c1);
    c2 = wmma_f16(af, b2, c2);
    c3 = wmma_f16(af, b3, c3);
  }
#pragma unroll
  for (int r = 0; r < 8; ++r) {
    size_t row = (size_t)(m0 + r + 8 * half) * DIM;
    out[row + n0 +  0 + l15] = c0[r];
    out[row + n0 + 16 + l15] = c1[r];
    out[row + n0 + 32 + l15] = c2[r];
    out[row + n0 + 48 + l15] = c3[r];
  }
}

// ---------------------------------------------------------------------------
extern "C" void kernel_launch(void* const* d_in, const int* in_sizes, int n_in,
                              void* d_out, int out_size, void* d_ws, size_t ws_size,
                              hipStream_t stream) {
  const float* x  = (const float*)d_in[0];
  const float* wq = (const float*)d_in[1];
  const float* wk = (const float*)d_in[2];
  const float* wv = (const float*)d_in[3];
  const float* wo = (const float*)d_in[4];
  float* outp = (float*)d_out;

  const size_t MB = 1024 * 1024;
  char* ws = (char*)d_ws;
  _Float16* x16  = (_Float16*)(ws);             // 16 MB  (2*2048*2048)
  _Float16* wq16 = (_Float16*)(ws + 16 * MB);   //  8 MB  (2048*2048)
  _Float16* wk16 = (_Float16*)(ws + 24 * MB);   //  2 MB  ( 512*2048)
  _Float16* wv16 = (_Float16*)(ws + 26 * MB);   //  2 MB
  _Float16* wo16 = (_Float16*)(ws + 28 * MB);   //  8 MB
  _Float16* qbuf = (_Float16*)(ws + 36 * MB);   // 16 MB  [B][32][S][64]
  _Float16* kbuf = (_Float16*)(ws + 52 * MB);   //  4 MB  [B][8][S][64]
  _Float16* vbuf = (_Float16*)(ws + 56 * MB);   //  4 MB
  _Float16* abuf = (_Float16*)(ws + 60 * MB);   // 16 MB  [B][S][2048]

  // One-time precision conversion (elementwise, HBM-bandwidth bound).
  const int NX = 2 * S_LEN * DIM, NQ = DIM * DIM, NKW = (NKV * HD) * DIM;
  cvt_f16_kernel<<<NX  / (8 * 256), 256, 0, stream>>>(x,  x16,  NX);
  cvt_f16_kernel<<<NQ  / (8 * 256), 256, 0, stream>>>(wq, wq16, NQ);
  cvt_f16_kernel<<<NKW / (8 * 256), 256, 0, stream>>>(wk, wk16, NKW);
  cvt_f16_kernel<<<NKW / (8 * 256), 256, 0, stream>>>(wv, wv16, NKW);
  cvt_f16_kernel<<<NQ  / (8 * 256), 256, 0, stream>>>(wo, wo16, NQ);

  dim3 blk(128);
  // Q: 1/sqrt(64) folded in; K: rope, no scale; V: passthrough.
  qkv_rope_kernel<<<dim3(64, NH),  blk, 0, stream>>>(x16, wq16, qbuf, NH,  1, 0.125f);
  qkv_rope_kernel<<<dim3(64, NKV), blk, 0, stream>>>(x16, wk16, kbuf, NKV, 1, 1.0f);
  qkv_rope_kernel<<<dim3(64, NKV), blk, 0, stream>>>(x16, wv16, vbuf, NKV, 0, 1.0f);
  attn_kernel<<<dim3(S_LEN / 64, NH, 2), blk, 0, stream>>>(qbuf, kbuf, vbuf, abuf);
  oproj_kernel<<<dim3(DIM / 64, 4096 / 64), blk, 0, stream>>>(abuf, wo16, outp);
}
